// SingleHeadAttention_56822417326257
// MI455X (gfx1250) — matrix operand order repaired and checked by reading, not compiled, not run on hardware
//
#include <hip/hip_runtime.h>
#include <hip/hip_bf16.h>
#include <math.h>

// ---------------------------------------------------------------------------
// SingleHeadAttention on MI455X (gfx1250).
// f16 WMMA pipeline, LDS double-buffered 128x128 block GEMM,
// async global->LDS staging (ASYNCcnt), f32 softmax.
// ---------------------------------------------------------------------------

#define DIM   768
#define BB    8
#define NN    2048
#define E3    (3 * DIM)          // 2304
#define MTOT  (BB * NN)          // 16384 flat token rows
#define TEMP  3.0f

#ifndef USE_ASYNC
#define USE_ASYNC 1              // gfx1250 async global->LDS staging
#endif

#define LDS_PAD 40               // halfs per staged row (32 data + 8 pad)

typedef __attribute__((ext_vector_type(16))) _Float16 v16h;
typedef __attribute__((ext_vector_type(8)))  _Float16 v8h;
typedef __attribute__((ext_vector_type(8)))  float    v8f;

#define WMMA_F16(a, b, c) \
  __builtin_amdgcn_wmma_f32_16x16x32_f16(false, (a), false, (b), (short)0, (c), false, false)

// ---- async global -> LDS, 16B per lane (gfx1250, ASYNCcnt tracked) ---------
__device__ __forceinline__ void async_cp16(_Float16* lds, const _Float16* g) {
#if USE_ASYNC
  const unsigned loff = (unsigned)(uintptr_t)lds;   // low 32 bits = LDS offset
  asm volatile("global_load_async_to_lds_b128 %0, %1, off"
               :: "v"(loff), "v"(g) : "memory");
#else
  v8h t = *(const v8h*)g;
  *(v8h*)lds = t;
#endif
}

__device__ __forceinline__ void wait_async_lds() {
#if USE_ASYNC
  asm volatile("s_wait_asynccnt 0x0" ::: "memory");
#endif
}

// ---- WMMA fragment loads from LDS tiles (wave32 layouts, CDNA5 ISA) -------
// A fragment (16x32): lane<16 -> row=lane, K={0..7,16..23}; lane>=16 -> K+8.
__device__ __forceinline__ v16h lds_a_frag(const _Float16* tile, int row0, int lane) {
  const int half = lane >> 4;
  const int r    = lane & 15;
  const _Float16* p = tile + (row0 + r) * LDS_PAD + (half ? 8 : 0);
  v8h lo = *(const v8h*)p;
  v8h hi = *(const v8h*)(p + 16);
  v16h f;
#pragma unroll
  for (int i = 0; i < 8; ++i) { f[i] = lo[i]; f[i + 8] = hi[i]; }
  return f;
}

// B fragment (32x16, stored N x K): lane<16 -> col=lane, K={0..7,16..23}; lane>=16 -> K+8.
__device__ __forceinline__ v16h lds_b_frag(const _Float16* tile, int col0, int lane) {
  const int half = lane >> 4;
  const int c    = lane & 15;
  const _Float16* p = tile + (col0 + c) * LDS_PAD + (half ? 8 : 0);
  v8h lo = *(const v8h*)p;
  v8h hi = *(const v8h*)(p + 16);
  v16h f;
#pragma unroll
  for (int i = 0; i < 8; ++i) { f[i] = lo[i]; f[i + 8] = hi[i]; }
  return f;
}

// ---- cooperative staging of one 128x32 A-tile + 128x32 B-tile --------------
// 256 threads: thread t copies 32B of A row t/2 and 32B of B row t/2.
__device__ __forceinline__ void stage_issue(const _Float16* __restrict__ gA, int ldA,
                                            const _Float16* __restrict__ gB, int ldB,
                                            int k0, _Float16* As, _Float16* Bs, int tid) {
  const int row = tid >> 1;
  const int kh  = (tid & 1) * 16;
  const _Float16* ga = gA + (size_t)row * ldA + k0 + kh;
  const _Float16* gb = gB + (size_t)row * ldB + k0 + kh;
  _Float16* la = As + row * LDS_PAD + kh;
  _Float16* lb = Bs + row * LDS_PAD + kh;
  async_cp16(la,     ga);
  async_cp16(la + 8, ga + 8);
  async_cp16(lb,     gb);
  async_cp16(lb + 8, gb + 8);
}

// ---- 128x128 block GEMM core: wave computes 32x64 (2x4 tiles, 8 WMMA/step) -
__device__ __forceinline__ void gemm128_core(const _Float16* __restrict__ gA, int ldA,
                                             const _Float16* __restrict__ gB, int ldB,
                                             int K, v8f acc[2][4]) {
  __shared__ _Float16 As[2][128 * LDS_PAD];
  __shared__ _Float16 Bs[2][128 * LDS_PAD];

  const int tid  = threadIdx.x;
  const int lane = tid & 31;
  const int wave = tid >> 5;
  const int wr   = wave >> 1;        // 0..3 -> m offset 32*wr
  const int wc   = wave & 1;         // 0..1 -> n offset 64*wc
  const int S    = K / 32;

  stage_issue(gA, ldA, gB, ldB, 0, As[0], Bs[0], tid);
  wait_async_lds();
  __syncthreads();

#pragma unroll 2
  for (int s = 0; s < S; ++s) {
    const int b = s & 1;
    if (s + 1 < S)
      stage_issue(gA, ldA, gB, ldB, (s + 1) * 32, As[1 - b], Bs[1 - b], tid);

    v16h a0 = lds_a_frag(As[b], 32 * wr,      lane);
    v16h a1 = lds_a_frag(As[b], 32 * wr + 16, lane);
    v16h b0 = lds_b_frag(Bs[b], 64 * wc,      lane);
    v16h b1 = lds_b_frag(Bs[b], 64 * wc + 16, lane);
    v16h b2 = lds_b_frag(Bs[b], 64 * wc + 32, lane);
    v16h b3 = lds_b_frag(Bs[b], 64 * wc + 48, lane);

    acc[0][0] = WMMA_F16(a0, b0, acc[0][0]);
    acc[0][1] = WMMA_F16(a0, b1, acc[0][1]);
    acc[0][2] = WMMA_F16(a0, b2, acc[0][2]);
    acc[0][3] = WMMA_F16(a0, b3, acc[0][3]);
    acc[1][0] = WMMA_F16(a1, b0, acc[1][0]);
    acc[1][1] = WMMA_F16(a1, b1, acc[1][1]);
    acc[1][2] = WMMA_F16(a1, b2, acc[1][2]);
    acc[1][3] = WMMA_F16(a1, b3, acc[1][3]);

    wait_async_lds();
    __syncthreads();
  }
}

// ---------------------------------------------------------------------------
// Kernel 0: f32 -> f16 convert (grid-stride)
// ---------------------------------------------------------------------------
__global__ __launch_bounds__(256) void cvt_f32_to_f16(const float* __restrict__ src,
                                                      _Float16* __restrict__ dst,
                                                      int count) {
  for (int i = blockIdx.x * 256 + threadIdx.x; i < count; i += gridDim.x * 256)
    dst[i] = (_Float16)src[i];
}

// ---------------------------------------------------------------------------
// Kernel 1: QKV projection. Block = 128 rows x 128 outs. Routes to q/k/vT.
// ---------------------------------------------------------------------------
__global__ __launch_bounds__(256) void qkv_gemm(const _Float16* __restrict__ xh,
                                                const _Float16* __restrict__ wh,
                                                const float* __restrict__ bias,
                                                _Float16* __restrict__ q,
                                                _Float16* __restrict__ k,
                                                _Float16* __restrict__ vT,
                                                float qscale) {
  const int nbx = E3 / 128;                        // 18
  const int gm0 = (blockIdx.x / nbx) * 128;
  const int gn0 = (blockIdx.x % nbx) * 128;

  v8f acc[2][4] = {};
  gemm128_core(xh + (size_t)gm0 * DIM, DIM, wh + (size_t)gn0 * DIM, DIM, DIM, acc);

  const int lane = threadIdx.x & 31;
  const int wave = threadIdx.x >> 5;
  const int wr = wave >> 1, wc = wave & 1;
  const int n = lane & 15, half = lane >> 4;

#pragma unroll
  for (int tm = 0; tm < 2; ++tm) {
#pragma unroll
    for (int tn = 0; tn < 4; ++tn) {
      const int row0 = gm0 + 32 * wr + 16 * tm;
      const int e0   = gn0 + 64 * wc + 16 * tn;
      const float bv = bias[e0 + n];
      const v8f c = acc[tm][tn];
      if (e0 < DIM) {                              // Q (scale folded in)
        _Float16* dst = q + (size_t)row0 * DIM + e0;
#pragma unroll
        for (int r = 0; r < 8; ++r)
          dst[(size_t)(r + 8 * half) * DIM + n] = (_Float16)((c[r] + bv) * qscale);
      } else if (e0 < 2 * DIM) {                   // K
        _Float16* dst = k + (size_t)row0 * DIM + (e0 - DIM);
#pragma unroll
        for (int r = 0; r < 8; ++r)
          dst[(size_t)(r + 8 * half) * DIM + n] = (_Float16)(c[r] + bv);
      } else {                                     // V -> transposed [B][DIM][NN]
        const int bidx = row0 >> 11;
        const int tok0 = row0 & (NN - 1);
        _Float16* dst = vT + ((size_t)bidx * DIM + (e0 - 2 * DIM) + n) * NN + tok0;
#pragma unroll
        for (int r = 0; r < 8; ++r)
          dst[r + 8 * half] = (_Float16)(c[r] + bv);
      }
    }
  }
}

// ---------------------------------------------------------------------------
// Kernel 2: logits S[b][i][j] = q[b][i][:] . k[b][j][:]   (scale in q)
// ---------------------------------------------------------------------------
__global__ __launch_bounds__(256) void scores_gemm(const _Float16* __restrict__ q,
                                                   const _Float16* __restrict__ k,
                                                   _Float16* __restrict__ s) {
  const int b  = blockIdx.x >> 8;                  // 256 blocks per batch
  const int t  = blockIdx.x & 255;
  const int gm0 = (t >> 4) * 128;
  const int gn0 = (t & 15) * 128;

  v8f acc[2][4] = {};
  gemm128_core(q + ((size_t)b * NN + gm0) * DIM, DIM,
               k + ((size_t)b * NN + gn0) * DIM, DIM, DIM, acc);

  const int lane = threadIdx.x & 31;
  const int wave = threadIdx.x >> 5;
  const int wr = wave >> 1, wc = wave & 1;
  const int n = lane & 15, half = lane >> 4;

#pragma unroll
  for (int tm = 0; tm < 2; ++tm) {
#pragma unroll
    for (int tn = 0; tn < 4; ++tn) {
      const int row0 = gm0 + 32 * wr + 16 * tm;
      const int j0   = gn0 + 64 * wc + 16 * tn;
      _Float16* dst = s + ((size_t)b * NN + row0) * NN + j0;
      const v8f c = acc[tm][tn];
#pragma unroll
      for (int r = 0; r < 8; ++r)
        dst[(size_t)(r + 8 * half) * NN + n] = (_Float16)c[r];
    }
  }
}

// ---------------------------------------------------------------------------
// Kernel 3: in-place row softmax (f32 math, f16 storage).
// ---------------------------------------------------------------------------
__global__ __launch_bounds__(256) void softmax_rows(_Float16* __restrict__ s) {
  const size_t row = blockIdx.x;                   // [0, BB*NN)
  _Float16* p = s + row * (size_t)NN;
  const int t = threadIdx.x;

  float v[8];
  float m = -1e30f;
#pragma unroll
  for (int i = 0; i < 8; ++i) {
    v[i] = (float)p[t + 256 * i];
    m = fmaxf(m, v[i]);
  }

  __shared__ float red[256];
  red[t] = m;
  __syncthreads();
#pragma unroll
  for (int off = 128; off > 0; off >>= 1) {
    if (t < off) red[t] = fmaxf(red[t], red[t + off]);
    __syncthreads();
  }
  m = red[0];
  __syncthreads();

  float sum = 0.f;
#pragma unroll
  for (int i = 0; i < 8; ++i) {
    v[i] = __expf(v[i] - m);
    sum += v[i];
  }
  red[t] = sum;
  __syncthreads();
#pragma unroll
  for (int off = 128; off > 0; off >>= 1) {
    if (t < off) red[t] += red[t + off];
    __syncthreads();
  }
  const float inv = 1.0f / red[0];
  __syncthreads();

#pragma unroll
  for (int i = 0; i < 8; ++i)
    p[t + 256 * i] = (_Float16)(v[i] * inv);
}

// ---------------------------------------------------------------------------
// Kernel 4: O[b][i][d] = sum_m P[b][i][m] * V[b][m][d]   (vT = [B][DIM][NN])
// ---------------------------------------------------------------------------
__global__ __launch_bounds__(256) void av_gemm(const _Float16* __restrict__ pr,
                                               const _Float16* __restrict__ vT,
                                               _Float16* __restrict__ o) {
  const int per_b = (NN / 128) * (DIM / 128);      // 96
  const int b  = blockIdx.x / per_b;
  const int t  = blockIdx.x % per_b;
  const int gm0 = (t / (DIM / 128)) * 128;
  const int gn0 = (t % (DIM / 128)) * 128;

  v8f acc[2][4] = {};
  gemm128_core(pr + ((size_t)b * NN + gm0) * NN, NN,
               vT + ((size_t)b * DIM + gn0) * NN, NN, NN, acc);

  const int lane = threadIdx.x & 31;
  const int wave = threadIdx.x >> 5;
  const int wr = wave >> 1, wc = wave & 1;
  const int n = lane & 15, half = lane >> 4;

#pragma unroll
  for (int tm = 0; tm < 2; ++tm) {
#pragma unroll
    for (int tn = 0; tn < 4; ++tn) {
      const int row0 = gm0 + 32 * wr + 16 * tm;
      const int d0   = gn0 + 64 * wc + 16 * tn;
      _Float16* dst = o + ((size_t)b * NN + row0) * DIM + d0;
      const v8f c = acc[tm][tn];
#pragma unroll
      for (int r = 0; r < 8; ++r)
        dst[(size_t)(r + 8 * half) * DIM + n] = (_Float16)c[r];
    }
  }
}

// ---------------------------------------------------------------------------
// Kernel 5: output projection, f32 stores to d_out.
// ---------------------------------------------------------------------------
__global__ __launch_bounds__(256) void proj_gemm(const _Float16* __restrict__ o,
                                                 const _Float16* __restrict__ wh,
                                                 const float* __restrict__ bias,
                                                 float* __restrict__ out) {
  const int nbx = DIM / 128;                       // 6
  const int gm0 = (blockIdx.x / nbx) * 128;
  const int gn0 = (blockIdx.x % nbx) * 128;

  v8f acc[2][4] = {};
  gemm128_core(o + (size_t)gm0 * DIM, DIM, wh + (size_t)gn0 * DIM, DIM, DIM, acc);

  const int lane = threadIdx.x & 31;
  const int wave = threadIdx.x >> 5;
  const int wr = wave >> 1, wc = wave & 1;
  const int n = lane & 15, half = lane >> 4;

#pragma unroll
  for (int tm = 0; tm < 2; ++tm) {
#pragma unroll
    for (int tn = 0; tn < 4; ++tn) {
      const int row0 = gm0 + 32 * wr + 16 * tm;
      const int e0   = gn0 + 64 * wc + 16 * tn;
      const float bv = bias[e0 + n];
      float* dst = out + (size_t)row0 * DIM + e0;
      const v8f c = acc[tm][tn];
#pragma unroll
      for (int r = 0; r < 8; ++r)
        dst[(size_t)(r + 8 * half) * DIM + n] = c[r] + bv;
    }
  }
}

// ---------------------------------------------------------------------------
// Host-side launcher
// ---------------------------------------------------------------------------
extern "C" void kernel_launch(void* const* d_in, const int* in_sizes, int n_in,
                              void* d_out, int out_size, void* d_ws, size_t ws_size,
                              hipStream_t stream) {
  const float* x      = (const float*)d_in[0];   // [B,N,DIM]
  const float* w_qkv  = (const float*)d_in[1];   // [3*DIM, DIM]
  const float* b_qkv  = (const float*)d_in[2];   // [3*DIM]
  const float* w_proj = (const float*)d_in[3];   // [DIM, DIM]
  const float* b_proj = (const float*)d_in[4];   // [DIM]
  float* out = (float*)d_out;                    // [B,N,DIM] f32

  _Float16* ws = (_Float16*)d_ws;
  const size_t n_x  = (size_t)MTOT * DIM;        // 12.58M halfs
  const size_t n_w3 = (size_t)E3 * DIM;
  const size_t n_wp = (size_t)DIM * DIM;
  const size_t n_s  = (size_t)BB * NN * NN;      // 33.55M halfs

  _Float16* xh     = ws;
  _Float16* wqkvh  = xh + n_x;
  _Float16* wprojh = wqkvh + n_w3;
  _Float16* q      = wprojh + n_wp;
  _Float16* k      = q + n_x;
  _Float16* vT     = k + n_x;
  _Float16* s      = vT + n_x;
  _Float16* o      = s + n_s;

  const float qscale = (1.0f / sqrtf((float)DIM)) / TEMP;

  // 0) convert inputs to f16
  cvt_f32_to_f16<<<4096, 256, 0, stream>>>(x, xh, (int)n_x);
  cvt_f32_to_f16<<<2048, 256, 0, stream>>>(w_qkv, wqkvh, (int)n_w3);
  cvt_f32_to_f16<<<1024, 256, 0, stream>>>(w_proj, wprojh, (int)n_wp);

  // 1) QKV projection: (16384/128)*(2304/128) = 2304 blocks
  qkv_gemm<<<2304, 256, 0, stream>>>(xh, wqkvh, b_qkv, q, k, vT, qscale);

  // 2) logits: 8 * (2048/128)^2 = 2048 blocks
  scores_gemm<<<2048, 256, 0, stream>>>(q, k, s);

  // 3) softmax over BB*NN rows
  softmax_rows<<<BB * NN, 256, 0, stream>>>(s);

  // 4) P*V: 8 * 16 * 6 = 768 blocks
  av_gemm<<<768, 256, 0, stream>>>(s, vT, o);

  // 5) output projection: 128 * 6 = 768 blocks
  proj_gemm<<<768, 256, 0, stream>>>(o, wprojh, b_proj, out);
}
